// Model_76725295776051
// MI455X (gfx1250) — matrix operand order repaired and checked by reading, not compiled, not run on hardware
//
#include <hip/hip_runtime.h>
#include <hip/hip_bf16.h>

// ---------------------------------------------------------------------------
// 2-layer tanh RNN (B=2048, H=1024, L=96, V=96) on CDNA5 via f16 WMMA.
// All GEMMs use v_wmma_f32_16x16x32_f16 (f16 operands, f32 accumulate).
// Weights converted once to f16 in workspace; hidden state ping-pongs between
// two f16 buffers. Epilogues (bias + rank-1 input term + tanh) in f32.
// RNN GEMMs: 32x64 wave tile (c[2][4]) -> 12 b128 loads per 8 WMMAs.
// Final sequential argmax patch: one wave32 per batch row.
// ---------------------------------------------------------------------------

typedef __attribute__((ext_vector_type(16))) _Float16 v16h;
typedef __attribute__((ext_vector_type(8)))  _Float16 v8h;
typedef __attribute__((ext_vector_type(8)))  float    v8f;

#define B_SZ 2048
#define H_SZ 1024
#define L_SZ 96
#define V_SZ 96

__device__ __forceinline__ v16h cat16(v8h lo, v8h hi) {
  return __builtin_shufflevector(lo, hi, 0, 1, 2, 3, 4, 5, 6, 7,
                                          8, 9, 10, 11, 12, 13, 14, 15);
}

// A-fragment (16x32, f16) from row-major A[M x lda], tile origin (m0, k0).
// Layout used here (contiguous K, not the ISA table order): lanes 0-15 -> M rows; per lane 16 contiguous K values
// K = k0 + half*16 + [0..15] (matches load_frag_bT).
__device__ __forceinline__ v16h load_frag_a(const _Float16* __restrict__ A,
                                            int lda, int m0, int k0) {
  int lane = threadIdx.x & 31;
  const _Float16* p = A + (size_t)(m0 + (lane & 15)) * lda
                        + (k0 + ((lane >> 4) << 4));
  v8h lo = *(const v8h*)(p);        // K = k0 + half*16 .. +7
  v8h hi = *(const v8h*)(p + 8);    // K = k0 + half*16 + 8 .. +15
  return cat16(lo, hi);
}

// B-fragment (32x16, f16) for C = A * W^T with W row-major [N x ldw].
// B[k][n] = W[n*ldw + k]; lane holds column n = n0 + lane%16,
// 16 contiguous K values starting at k0 + (lane/16)*16.
__device__ __forceinline__ v16h load_frag_bT(const _Float16* __restrict__ W,
                                             int ldw, int k0, int n0) {
  int lane = threadIdx.x & 31;
  const _Float16* p = W + (size_t)(n0 + (lane & 15)) * ldw
                        + (k0 + ((lane >> 4) << 4));
  v8h lo = *(const v8h*)(p);
  v8h hi = *(const v8h*)(p + 8);
  return cat16(lo, hi);
}

__device__ __forceinline__ v8f wmma_f16(v16h a, v16h b, v8f c) {
  // (neg_a, A, neg_b, B, c_mod, C, reuse_a, reuse_b)
  return __builtin_amdgcn_wmma_f32_16x16x32_f16(false, a, false, b,
                                                (short)0, c, false, false);
}

// Accumulate a 32(M) x 64(N) wave tile over one K-sweep of C += A * W^T.
__device__ __forceinline__ void gemm_sweep_32x64(
    v8f c[2][4], const _Float16* __restrict__ A, const _Float16* __restrict__ W,
    int m0, int n0) {
  for (int k0 = 0; k0 < H_SZ; k0 += 32) {
    v16h a0 = load_frag_a(A, H_SZ, m0,      k0);
    v16h a1 = load_frag_a(A, H_SZ, m0 + 16, k0);
#pragma unroll
    for (int ni = 0; ni < 4; ++ni) {
      v16h b = load_frag_bT(W, H_SZ, k0, n0 + ni * 16);
      c[0][ni] = wmma_f16(a0, b, c[0][ni]);
      c[1][ni] = wmma_f16(a1, b, c[1][ni]);
    }
  }
}

// ---------------------------------------------------------------------------
// Layer 0: h0_new = tanh(x_t * Wih + bih + bhh + h0_old @ Whh^T)
// Block 256 threads = 8 waves arranged 4(M) x 2(N); block tile 128 x 128.
// ---------------------------------------------------------------------------
__global__ __launch_bounds__(256) void rnn_l0_kernel(
    const _Float16* __restrict__ hprev,   // [B, H] f16
    const _Float16* __restrict__ whh,     // [H, H] f16 (row-major, as given)
    const float* __restrict__ x,          // [B, 1, L] f32
    const float* __restrict__ wih,        // [H] f32
    const float* __restrict__ bih,
    const float* __restrict__ bhh,
    _Float16* __restrict__ hout,          // [B, H] f16
    float* __restrict__ hfinal,           // nullptr except last step
    int t)
{
  int wave = threadIdx.x >> 5;
  int lane = threadIdx.x & 31;
  int m0 = blockIdx.x * 128 + (wave >> 1) * 32;
  int n0 = blockIdx.y * 128 + (wave & 1) * 64;

  v8f c[2][4] = {};
  gemm_sweep_32x64(c, hprev, whh, m0, n0);

  for (int mi = 0; mi < 2; ++mi) {
    for (int ni = 0; ni < 4; ++ni) {
      int n  = n0 + ni * 16 + (lane & 15);
      int mb = m0 + mi * 16 + (lane >> 4) * 8;
      float wv = wih[n];
      float bb = bih[n] + bhh[n];
      for (int r = 0; r < 8; ++r) {
        int m = mb + r;
        float acc = c[mi][ni][r] + x[(size_t)m * L_SZ + t] * wv + bb;
        float h = tanhf(acc);
        hout[(size_t)m * H_SZ + n] = (_Float16)h;
        if (hfinal) hfinal[(size_t)m * H_SZ + n] = h;
      }
    }
  }
}

// ---------------------------------------------------------------------------
// Layer 1: h1_new = tanh(h0_new @ Wih^T + bih + h1_old @ Whh^T + bhh)
// ---------------------------------------------------------------------------
__global__ __launch_bounds__(256) void rnn_l1_kernel(
    const _Float16* __restrict__ h0new,   // [B, H] f16
    const _Float16* __restrict__ w1ih,    // [H, H] f16
    const _Float16* __restrict__ h1old,   // [B, H] f16
    const _Float16* __restrict__ w1hh,    // [H, H] f16
    const float* __restrict__ bih,
    const float* __restrict__ bhh,
    _Float16* __restrict__ hout,
    float* __restrict__ hfinal)
{
  int wave = threadIdx.x >> 5;
  int lane = threadIdx.x & 31;
  int m0 = blockIdx.x * 128 + (wave >> 1) * 32;
  int n0 = blockIdx.y * 128 + (wave & 1) * 64;

  v8f c[2][4] = {};
  gemm_sweep_32x64(c, h0new, w1ih, m0, n0);
  gemm_sweep_32x64(c, h1old, w1hh, m0, n0);

  for (int mi = 0; mi < 2; ++mi) {
    for (int ni = 0; ni < 4; ++ni) {
      int n  = n0 + ni * 16 + (lane & 15);
      int mb = m0 + mi * 16 + (lane >> 4) * 8;
      float bb = bih[n] + bhh[n];
      for (int r = 0; r < 8; ++r) {
        int m = mb + r;
        float h = tanhf(c[mi][ni][r] + bb);
        hout[(size_t)m * H_SZ + n] = (_Float16)h;
        if (hfinal) hfinal[(size_t)m * H_SZ + n] = h;
      }
    }
  }
}

// ---------------------------------------------------------------------------
// FC: logits[:, t, :] = h1 @ fcW^T + fcb.  V=96 = 6 N-tiles of 16.
// Block = 96 threads (3 waves); wave handles n0 = wave*32; block covers 32 rows.
// ---------------------------------------------------------------------------
__global__ __launch_bounds__(96) void fc_kernel(
    const _Float16* __restrict__ h1,      // [B, H] f16
    const _Float16* __restrict__ fcw,     // [V, H] f16
    const float* __restrict__ fcb,        // [V]
    float* __restrict__ out,              // [B, L, V] f32
    int t)
{
  int wave = threadIdx.x >> 5;
  int lane = threadIdx.x & 31;
  int m0 = blockIdx.x * 32;
  int n0 = wave * 32;

  v8f c[2][2] = {};
  for (int k0 = 0; k0 < H_SZ; k0 += 32) {
    v16h a0 = load_frag_a(h1, H_SZ, m0,      k0);
    v16h a1 = load_frag_a(h1, H_SZ, m0 + 16, k0);
    v16h b0 = load_frag_bT(fcw, H_SZ, k0, n0);
    v16h b1 = load_frag_bT(fcw, H_SZ, k0, n0 + 16);
    c[0][0] = wmma_f16(a0, b0, c[0][0]);
    c[0][1] = wmma_f16(a0, b1, c[0][1]);
    c[1][0] = wmma_f16(a1, b0, c[1][0]);
    c[1][1] = wmma_f16(a1, b1, c[1][1]);
  }

  for (int mi = 0; mi < 2; ++mi) {
    for (int ni = 0; ni < 2; ++ni) {
      int n  = n0 + ni * 16 + (lane & 15);
      int mb = m0 + mi * 16 + (lane >> 4) * 8;
      float bb = fcb[n];
      for (int r = 0; r < 8; ++r) {
        int m = mb + r;
        out[(size_t)m * (L_SZ * V_SZ) + (size_t)t * V_SZ + n] = c[mi][ni][r] + bb;
      }
    }
  }
}

// ---------------------------------------------------------------------------
// Conversions / init
// ---------------------------------------------------------------------------
__global__ void f32_to_f16_kernel(const float* __restrict__ src,
                                  _Float16* __restrict__ dst, int n) {
  for (int i = blockIdx.x * blockDim.x + threadIdx.x; i < n;
       i += gridDim.x * blockDim.x)
    dst[i] = (_Float16)src[i];
}

__global__ void zero_f16_kernel(_Float16* __restrict__ dst, int n) {
  for (int i = blockIdx.x * blockDim.x + threadIdx.x; i < n;
       i += gridDim.x * blockDim.x)
    dst[i] = (_Float16)0.0f;
}

// ---------------------------------------------------------------------------
// Sequential probability patch. One wave32 per batch row; V=96 = 3 per lane.
// Matches jnp.argmax first-occurrence tie-breaking. Adjustments applied in
// registers (so the recomputed argmax needs no memory round-trip) and stored.
// ---------------------------------------------------------------------------
__device__ __forceinline__ int wave_argmax96(float v0, float v1, float v2,
                                             int lane) {
  float best = v0; int bi = lane;
  if (v1 > best) { best = v1; bi = lane + 32; }
  if (v2 > best) { best = v2; bi = lane + 64; }
  for (int off = 16; off > 0; off >>= 1) {
    float ov = __shfl_xor(best, off, 32);
    int   oi = __shfl_xor(bi,   off, 32);
    if (ov > best || (ov == best && oi < bi)) { best = ov; bi = oi; }
  }
  return bi;
}

__global__ __launch_bounds__(256) void adjust_kernel(float* __restrict__ out) {
  int wave = threadIdx.x >> 5;
  int lane = threadIdx.x & 31;
  int b = blockIdx.x * 8 + wave;            // grid sized exactly: b < 2048
  float* base = out + (size_t)b * (L_SZ * V_SZ);

  float v0 = base[lane], v1 = base[lane + 32], v2 = base[lane + 64];
  int prev_arg = wave_argmax96(v0, v1, v2, lane);   // row 0 stays unadjusted

  for (int t = 1; t < L_SZ; ++t) {
    float* row = base + t * V_SZ;
    v0 = row[lane]; v1 = row[lane + 32]; v2 = row[lane + 64];
    int cur_arg = wave_argmax96(v0, v1, v2, lane);  // argmax BEFORE adjustment
    bool last = (t == L_SZ - 1);
    if (prev_arg == 0 && cur_arg != 1) {            // uniform across wave
      if (lane == 1) { v0 += 0.5f; row[1] = v0; }
    }
    if (last && cur_arg == 0) {
      if (lane == 0) { v0 -= 0.5f; row[0] = v0; }
    }
    prev_arg = wave_argmax96(v0, v1, v2, lane);     // argmax AFTER adjustment
  }
}

// ---------------------------------------------------------------------------
// Host-side orchestration (all on `stream`; graph-capture safe).
// ---------------------------------------------------------------------------
extern "C" void kernel_launch(void* const* d_in, const int* in_sizes, int n_in,
                              void* d_out, int out_size, void* d_ws,
                              size_t ws_size, hipStream_t stream) {
  (void)in_sizes; (void)n_in; (void)out_size;

  const float* x    = (const float*)d_in[0];   // [B,1,L]
  const float* w0ih = (const float*)d_in[1];   // [H,1]
  const float* w0hh = (const float*)d_in[2];   // [H,H]
  const float* b0ih = (const float*)d_in[3];
  const float* b0hh = (const float*)d_in[4];
  const float* w1ih = (const float*)d_in[5];   // [H,H]
  const float* w1hh = (const float*)d_in[6];   // [H,H]
  const float* b1ih = (const float*)d_in[7];
  const float* b1hh = (const float*)d_in[8];
  const float* fcw  = (const float*)d_in[9];   // [V,H]
  const float* fcb  = (const float*)d_in[10];

  char* ws = (char*)d_ws;
  const size_t WB = (size_t)H_SZ * H_SZ * sizeof(_Float16);   // 2 MB
  const size_t FB = (size_t)V_SZ * H_SZ * sizeof(_Float16);   // 192 KB
  const size_t HB = (size_t)B_SZ * H_SZ * sizeof(_Float16);   // 4 MB
  _Float16* w0hh_h = (_Float16*)(ws);
  _Float16* w1ih_h = (_Float16*)(ws + WB);
  _Float16* w1hh_h = (_Float16*)(ws + 2 * WB);
  _Float16* fcw_h  = (_Float16*)(ws + 3 * WB);
  size_t off = 3 * WB + FB;
  _Float16* h0buf[2] = { (_Float16*)(ws + off),
                         (_Float16*)(ws + off + HB) };
  _Float16* h1buf[2] = { (_Float16*)(ws + off + 2 * HB),
                         (_Float16*)(ws + off + 3 * HB) };
  if (ws_size < off + 4 * HB) return;   // ~22.3 MB required

  f32_to_f16_kernel<<<512, 256, 0, stream>>>(w0hh, w0hh_h, H_SZ * H_SZ);
  f32_to_f16_kernel<<<512, 256, 0, stream>>>(w1ih, w1ih_h, H_SZ * H_SZ);
  f32_to_f16_kernel<<<512, 256, 0, stream>>>(w1hh, w1hh_h, H_SZ * H_SZ);
  f32_to_f16_kernel<<<128, 256, 0, stream>>>(fcw,  fcw_h,  V_SZ * H_SZ);
  zero_f16_kernel<<<512, 256, 0, stream>>>(h0buf[0], B_SZ * H_SZ);
  zero_f16_kernel<<<512, 256, 0, stream>>>(h1buf[0], B_SZ * H_SZ);

  float* outF = (float*)d_out;
  float* h0_final = outF + (size_t)B_SZ * L_SZ * V_SZ;      // hidden[0]
  float* h1_final = h0_final + (size_t)B_SZ * H_SZ;         // hidden[1]

  dim3 grid(B_SZ / 128, H_SZ / 128);  // 16 x 8 blocks, 256 thr (8 waves) each
  for (int t = 0; t < L_SZ; ++t) {
    int cur = t & 1, nxt = (t + 1) & 1;
    bool last = (t == L_SZ - 1);
    rnn_l0_kernel<<<grid, 256, 0, stream>>>(
        h0buf[cur], w0hh_h, x, w0ih, b0ih, b0hh,
        h0buf[nxt], last ? h0_final : nullptr, t);
    rnn_l1_kernel<<<grid, 256, 0, stream>>>(
        h0buf[nxt], w1ih_h, h1buf[cur], w1hh_h, b1ih, b1hh,
        h1buf[nxt], last ? h1_final : nullptr);
    fc_kernel<<<B_SZ / 32, 96, 0, stream>>>(h1buf[nxt], fcw_h, fcb, outF, t);
  }

  adjust_kernel<<<B_SZ / 8, 256, 0, stream>>>(outF);
}
